// Localizer_18279380812165
// MI455X (gfx1250) — matrix operand order repaired and checked by reading, not compiled, not run on hardware
//
#include <hip/hip_runtime.h>
#include <hip/hip_bf16.h>
#include <math.h>
#include <stdint.h>

#define P_N 128
#define B_N 32
#define S_N 100
#define H_N 128
#define PB  4096
#define XS  528           // xpart row stride (513 cols padded to 33 WMMA tiles)
#define LOG_INV_P -4.852030263919617f   // log(1/128)

typedef _Float16 v16h __attribute__((ext_vector_type(16)));
typedef _Float16 v8h  __attribute__((ext_vector_type(8)));
typedef _Float16 v4h  __attribute__((ext_vector_type(4)));
typedef float    v8f  __attribute__((ext_vector_type(8)));

union FragU { v16h v; struct { v8h lo, hi; } s; };

// ---------- small device helpers ----------
__device__ __forceinline__ unsigned hash_u32(unsigned x) {
  x ^= x >> 16; x *= 0x7feb352dU; x ^= x >> 15; x *= 0x846ca68bU; x ^= x >> 16;
  return x;
}
__device__ __forceinline__ float u01(unsigned x) {
  return ((float)hash_u32(x) + 0.5f) * 2.3283064365386963e-10f;
}
__device__ __forceinline__ float nrand(unsigned s) {
  float u1 = u01(s * 2u + 0x9E3779B9u);
  float u2 = u01(s * 2u + 0x85EBCA6Bu);
  float r  = sqrtf(-2.0f * logf(u1));
  return r * cosf(6.28318530717958647f * u2);
}
__device__ __forceinline__ float sigm(float x)  { return 1.0f / (1.0f + expf(-x)); }
__device__ __forceinline__ float softp(float x) { return (x > 20.f) ? x : log1pf(expf(x)); }

// ---------- WMMA fragment loaders (wave32, 16x16x32 f16) ----------
// A: 16xK tile, LDS row-major with stride 136 halfs (272B, 16B aligned, bank-friendly).
__device__ __forceinline__ v16h loadA(const _Float16* tile, int m0, int k0) {
  int lane = threadIdx.x & 31;
  int m  = m0 + (lane & 15);
  int kh = k0 + ((lane >> 4) << 3);
  const _Float16* p = tile + m * 136 + kh;
  FragU f;
  f.s.lo = *(const v8h*)(p);
  f.s.hi = *(const v8h*)(p + 16);
  return f.v;
}
// B: 32x16 tile of W^T [K][N], lane = N column (lane%16), elements = K rows 8h+s / 16+8h+s.
// Streams straight from global (weights broadcast across blocks -> resident in L2).
__device__ __forceinline__ v16h loadB(const _Float16* w, int k0, int n0, int stride) {
  int lane = threadIdx.x & 31;
  const _Float16* p = w + (size_t)(k0 + ((lane >> 4) << 3)) * stride + n0 + (lane & 15);
  v16h f;
#pragma unroll
  for (int s = 0; s < 8; s++) {
    f[s] = p[(size_t)s * stride];
    f[8 + s] = p[(size_t)(16 + s) * stride];
  }
  return f;
}

// Async-copy a 128x128 f16 tile (global, row stride 128) into LDS (row stride 136)
// using per-lane GLOBAL_LOAD_ASYNC_TO_LDS_B128 (ASYNCcnt path; no VGPR round-trip).
__device__ __forceinline__ void async_tile_load(_Float16* lds, const _Float16* __restrict__ src,
                                                int r0, int tid) {
  for (int i = tid; i < 128 * 16; i += 256) {
    int row = i >> 4, c8 = (i & 15) << 3;           // 8 halfs (16B) per transfer
    unsigned laddr = (unsigned)(uintptr_t)(lds + row * 136 + c8);
    unsigned long long gaddr =
        (unsigned long long)(uintptr_t)(src + (size_t)(r0 + row) * 128 + c8);
    asm volatile("global_load_async_to_lds_b128 %0, %1, off"
                 :: "v"(laddr), "v"(gaddr) : "memory");
  }
  asm volatile("s_wait_asynccnt 0x0" ::: "memory");
}

// ---------- weight prep: transpose/convert to f16 K-major layouts ----------
__global__ void k_prep(const float* __restrict__ fcz_w, const float* __restrict__ fcz_b,
                       const float* __restrict__ fcr_w, const float* __restrict__ fcr_b,
                       const float* __restrict__ fcn_w, const float* __restrict__ fcn_b,
                       const float* __restrict__ fco_w, const float* __restrict__ fco_b,
                       _Float16* __restrict__ Wzh, _Float16* __restrict__ Wrh,
                       _Float16* __restrict__ Wna, float* __restrict__ Wo,
                       _Float16* __restrict__ Wx, float* __restrict__ xbias) {
  int idx = blockIdx.x * 256 + threadIdx.x;
  if (idx < 128 * 128) {
    int k = idx >> 7, n = idx & 127;
    Wzh[k * 128 + n] = (_Float16)fcz_w[n * 256 + 128 + k];   // h-part of fcz
    Wrh[k * 128 + n] = (_Float16)fcr_w[n * 256 + 128 + k];   // h-part of fcr
  }
  if (idx < 128 * 256) {
    int k = idx >> 8, n = idx & 255;
    Wna[k * 256 + n] = (_Float16)fcn_w[n * 256 + k];         // (r*h)-part of fcn
  }
  if (idx < 128) Wo[idx] = fco_w[idx];                       // h1-part of fco
  if (idx < 128 * XS) {
    int k = idx / XS, n = idx % XS;
    float v = 0.f;
    if      (n < 128) v = fcz_w[n * 256 + k];                // x-part of fcz
    else if (n < 256) v = fcr_w[(n - 128) * 256 + k];        // x-part of fcr
    else if (n < 512) v = fcn_w[(n - 256) * 256 + 128 + k];  // x-part of fcn
    else if (n == 512) v = fco_w[128 + k];                   // x-part of fco
    Wx[(size_t)k * XS + n] = (_Float16)v;
  }
  if (idx < XS) {
    float v = 0.f;
    if      (idx < 128) v = fcz_b[idx];
    else if (idx < 256) v = fcr_b[idx - 128];
    else if (idx < 512) v = fcn_b[idx - 256];
    else if (idx == 512) v = fco_b[0];
    xbias[idx] = v;
  }
}

// ---------- map CNN encoder ----------
__global__ void k_conv1(const float* __restrict__ in, const float* __restrict__ w,
                        float* __restrict__ out) {
  int idx = blockIdx.x * 256 + threadIdx.x;
  if (idx >= 32 * 16 * 32 * 32) return;
  int x = idx & 31, y = (idx >> 5) & 31, c = (idx >> 10) & 15, b = idx >> 14;
  float acc = 0.f;
  for (int ky = 0; ky < 5; ky++) {
    int iy = y * 2 - 2 + ky;
    if (iy < 0 || iy >= 64) continue;
    for (int kx = 0; kx < 5; kx++) {
      int ix = x * 2 - 2 + kx;
      if (ix < 0 || ix >= 64) continue;
      acc += in[((size_t)b * 64 + iy) * 64 + ix] * w[c * 25 + ky * 5 + kx];
    }
  }
  out[idx] = fmaxf(acc, 0.f);
}

__global__ void k_conv3x3(const float* __restrict__ in, const float* __restrict__ w,
                          float* __restrict__ out, int Cin, int Cout) {
  int idx = blockIdx.x * 256 + threadIdx.x;
  if (idx >= 32 * Cout * 1024) return;
  int x = idx & 31, y = (idx >> 5) & 31, rest = idx >> 10;
  int c = rest % Cout, b = rest / Cout;
  float acc = 0.f;
  for (int ci = 0; ci < Cin; ci++) {
    for (int ky = 0; ky < 3; ky++) {
      int iy = y - 1 + ky;
      if (iy < 0 || iy >= 32) continue;
      for (int kx = 0; kx < 3; kx++) {
        int ix = x - 1 + kx;
        if (ix < 0 || ix >= 32) continue;
        acc += in[(((size_t)b * Cin + ci) * 32 + iy) * 32 + ix] *
               w[((c * Cin + ci) * 3 + ky) * 3 + kx];
      }
    }
  }
  out[idx] = fmaxf(acc, 0.f);
}

__global__ void k_bnstat(const float* __restrict__ x, float* __restrict__ stats, int C) {
  __shared__ float s1[256], s2[256];
  int c = blockIdx.x;
  float a = 0.f, b2 = 0.f;
  for (int i = threadIdx.x; i < 32 * 1024; i += 256) {
    int bb = i >> 10, j = i & 1023;
    float v = x[(((size_t)bb * C) + c) * 1024 + j];
    a += v; b2 += v * v;
  }
  s1[threadIdx.x] = a; s2[threadIdx.x] = b2;
  __syncthreads();
  for (int st = 128; st > 0; st >>= 1) {
    if (threadIdx.x < st) { s1[threadIdx.x] += s1[threadIdx.x + st]; s2[threadIdx.x] += s2[threadIdx.x + st]; }
    __syncthreads();
  }
  if (threadIdx.x == 0) {
    float m = s1[0] / 32768.f;
    float v = s2[0] / 32768.f - m * m;
    stats[c * 2] = m;
    stats[c * 2 + 1] = rsqrtf(v + 1e-5f);
  }
}

__global__ void k_bnapply(float* __restrict__ x, const float* __restrict__ stats,
                          const float* __restrict__ g, const float* __restrict__ bta, int C) {
  int idx = blockIdx.x * 256 + threadIdx.x;
  if (idx >= 32 * C * 1024) return;
  int c = (idx >> 10) % C;
  x[idx] = (x[idx] - stats[c * 2]) * stats[c * 2 + 1] * g[c] + bta[c];
}

__global__ void k_maplin(const float* __restrict__ x3, const float* __restrict__ mw,
                         const float* __restrict__ mb, float* __restrict__ embmap) {
  __shared__ float s[256];
  int b = blockIdx.x >> 7, o = blockIdx.x & 127;
  float acc = 0.f;
  for (int f = threadIdx.x; f < 32768; f += 256)
    acc += x3[(size_t)b * 32768 + f] * mw[(size_t)o * 32768 + f];
  s[threadIdx.x] = acc;
  __syncthreads();
  for (int st = 128; st > 0; st >>= 1) {
    if (threadIdx.x < st) s[threadIdx.x] += s[threadIdx.x + st];
    __syncthreads();
  }
  if (threadIdx.x == 0) embmap[b * 128 + o] = fmaxf(s[0] + mb[o], 0.f);
}

__global__ void k_m2(const float* __restrict__ embmap,
                     const float* __restrict__ m2o_w, const float* __restrict__ m2o_b,
                     const float* __restrict__ m2a_w, const float* __restrict__ m2a_b,
                     float* __restrict__ obsmap, float* __restrict__ actmap) {
  int b = blockIdx.x, j = threadIdx.x;
  const float* e = embmap + b * 128;
  if (j < 64) {
    float a = m2o_b[j];
    for (int k = 0; k < 128; k++) a += e[k] * m2o_w[j * 128 + k];
    obsmap[b * 64 + j] = fmaxf(a, 0.f);
  } else {
    int j2 = j - 64;
    float a = m2a_b[j2];
    for (int k = 0; k < 128; k++) a += e[k] * m2a_w[j2 * 128 + k];
    actmap[b * 64 + j2] = fmaxf(a, 0.f);
  }
}

// emb[s*32+b][0..127] f16  (obs embedding * obs_map | act embedding * act_map)
__global__ void k_emb(const float* __restrict__ obs_in, const float* __restrict__ act_in,
                      const float* __restrict__ obs_w, const float* __restrict__ obs_b,
                      const float* __restrict__ act_w, const float* __restrict__ act_b,
                      const float* __restrict__ obsmap, const float* __restrict__ actmap,
                      _Float16* __restrict__ emb) {
  int idx = blockIdx.x * 256 + threadIdx.x;
  if (idx >= 3200 * 128) return;
  int r = idx >> 7, j = idx & 127;
  int s = r >> 5, b = r & 31;
  float v;
  if (j < 64) {
    float a = obs_b[j];
    for (int k = 0; k < 16; k++) a += obs_in[((size_t)b * 100 + s) * 16 + k] * obs_w[j * 16 + k];
    v = fmaxf(a, 0.f) * obsmap[b * 64 + j];
  } else {
    int j2 = j - 64;
    float a = act_b[j2];
    for (int k = 0; k < 3; k++) a += act_in[((size_t)b * 100 + s) * 3 + k] * act_w[j2 * 3 + k];
    v = fmaxf(a, 0.f) * actmap[b * 64 + j2];
  }
  emb[(size_t)r * 128 + j] = (_Float16)v;
}

// xpart[3200][XS] = emb @ Wx + xbias (WMMA; all 100 steps of x-contributions at once)
__global__ void __launch_bounds__(256, 1)
k_xpart(const _Float16* __restrict__ emb, const _Float16* __restrict__ Wx,
        const float* __restrict__ xbias, float* __restrict__ xpart) {
  __shared__ __align__(16) _Float16 sA[128 * 136];
  int tid = threadIdx.x;
  int r0 = blockIdx.x * 128;
  async_tile_load(sA, emb, r0, tid);
  __syncthreads();
  int wv = tid >> 5, lane = tid & 31;
  int m0 = wv * 16, nl = lane & 15, mh = (lane >> 4) << 3;
  for (int jt = 0; jt < 33; jt++) {
    v8f acc = {};
    for (int kt = 0; kt < 4; kt++) {
      v16h a = loadA(sA, m0, kt * 32);
      v16h b = loadB(Wx, kt * 32, jt * 16, XS);
      acc = __builtin_amdgcn_wmma_f32_16x16x32_f16(false, a, false, b, (short)0, acc, false, false);
    }
    int n = jt * 16 + nl;
    float bias = xbias[n];
#pragma unroll
    for (int v = 0; v < 8; v++) {
      int m = m0 + mh + v;
      xpart[(size_t)(r0 + m) * XS + n] = acc[v] + bias;
    }
  }
}

__global__ void k_init(_Float16* __restrict__ h16, float* __restrict__ p) {
  int idx = blockIdx.x * 256 + threadIdx.x;
  if (idx < PB * H_N) h16[idx] = (_Float16)u01((unsigned)idx ^ 0xC0FFEEu);
  if (idx < PB) p[idx] = LOG_INV_P;
}

// ---------- recurrent step A: gates + candidate + h1 + logpdf (WMMA) ----------
__global__ void __launch_bounds__(256, 1)
k_stepA(int t, const _Float16* __restrict__ h16, const float* __restrict__ p,
        const float* __restrict__ xpart,
        const _Float16* __restrict__ Wzh, const _Float16* __restrict__ Wrh,
        const _Float16* __restrict__ Wna, const float* __restrict__ Wo,
        float* __restrict__ h1f, float* __restrict__ logpdf) {
  __shared__ __align__(16) _Float16 sH[128 * 136];
  __shared__ __align__(16) _Float16 sA[128 * 136];
  __shared__ __align__(16) _Float16 sH1[128 * 136];
  __shared__ float sWo[128];
  int tid = threadIdx.x;
  int r0 = blockIdx.x * 128;

  __builtin_prefetch(Wrh, 0, 1);
  __builtin_prefetch(Wzh, 0, 1);
  __builtin_prefetch(Wna, 0, 1);

  async_tile_load(sH, h16, r0, tid);          // ASYNCcnt memory->LDS copy of h tile
  if (tid < 128) sWo[tid] = Wo[tid];
  __syncthreads();

  int wv = tid >> 5, lane = tid & 31;
  int m0 = wv * 16, nl = lane & 15, mh = (lane >> 4) << 3;

  // phase 1: r-gate, build a = rg * h (each wave owns its own 16 rows of sA)
  for (int jt = 0; jt < 8; jt++) {
    v8f acc = {};
    for (int kt = 0; kt < 4; kt++) {
      v16h a = loadA(sH, m0, kt * 32);
      v16h b = loadB(Wrh, kt * 32, jt * 16, 128);
      acc = __builtin_amdgcn_wmma_f32_16x16x32_f16(false, a, false, b, (short)0, acc, false, false);
    }
    int n = jt * 16 + nl;
#pragma unroll
    for (int v = 0; v < 8; v++) {
      int m = m0 + mh + v;
      int grow = r0 + m;
      const float* xp = xpart + (size_t)(t * 32 + (grow & 31)) * XS;
      float rg = sigm(acc[v] + xp[128 + n]);
      sA[m * 136 + n] = (_Float16)(rg * (float)sH[m * 136 + n]);
    }
  }

  // phase 2: z-gate + (mu,var) + reparameterize + h1
  for (int jt = 0; jt < 8; jt++) {
    v8f zacc = {}; v8f macc = {}; v8f vacc = {};
    for (int kt = 0; kt < 4; kt++) {
      v16h aH = loadA(sH, m0, kt * 32);
      v16h aA = loadA(sA, m0, kt * 32);
      v16h bZ = loadB(Wzh, kt * 32, jt * 16, 128);
      v16h bM = loadB(Wna, kt * 32, jt * 16, 256);
      v16h bV = loadB(Wna, kt * 32, 128 + jt * 16, 256);
      zacc = __builtin_amdgcn_wmma_f32_16x16x32_f16(false, aH, false, bZ, (short)0, zacc, false, false);
      macc = __builtin_amdgcn_wmma_f32_16x16x32_f16(false, aA, false, bM, (short)0, macc, false, false);
      vacc = __builtin_amdgcn_wmma_f32_16x16x32_f16(false, aA, false, bV, (short)0, vacc, false, false);
    }
    int n = jt * 16 + nl;
#pragma unroll
    for (int v = 0; v < 8; v++) {
      int m = m0 + mh + v;
      int grow = r0 + m;
      const float* xp = xpart + (size_t)(t * 32 + (grow & 31)) * XS;
      float z   = sigm(zacc[v] + xp[n]);
      float mu  = macc[v] + xp[256 + n];
      float var = vacc[v] + xp[384 + n];
      float nn  = mu + nrand((unsigned)((t * PB + grow) * 128 + n)) * softp(var);
      float h1  = (1.f - z) * nn + z * (float)sH[m * 136 + n];
      sH1[m * 136 + n] = (_Float16)h1;
      h1f[(size_t)grow * 128 + n] = h1;
    }
  }
  __syncthreads();

  // phase 3: logpdf = h1 . Wo_h + x-part(+fco_b) + p
  if (tid < 128) {
    int m = tid, grow = r0 + m;
    float acc = 0.f;
#pragma unroll 8
    for (int k = 0; k < 128; k++) acc += (float)sH1[m * 136 + k] * sWo[k];
    const float* xp = xpart + (size_t)(t * 32 + (grow & 31)) * XS;
    logpdf[grow] = acc + xp[512] + p[grow];
  }
}

// ---------- step B: per-batch-column log-softmax + soft resampling ----------
__global__ void k_stepB(int t, const float* __restrict__ h1f, const float* __restrict__ logpdf,
                        _Float16* __restrict__ h16, float* __restrict__ p) {
  __shared__ float red[128];
  __shared__ float p1s[128];
  __shared__ float cdf[128];
  int b = blockIdx.x, i = threadIdx.x;
  int row = i * 32 + b;
  float lp = logpdf[row];

  red[i] = lp; __syncthreads();
  for (int st = 64; st > 0; st >>= 1) {
    if (i < st) red[i] = fmaxf(red[i], red[i + st]);
    __syncthreads();
  }
  float mx = red[0]; __syncthreads();
  float e = expf(lp - mx);
  red[i] = e; __syncthreads();
  for (int st = 64; st > 0; st >>= 1) {
    if (i < st) red[i] += red[i + st];
    __syncthreads();
  }
  float p1 = lp - (mx + logf(red[0])); __syncthreads();
  p1s[i] = p1;
  float res = 0.5f * expf(p1) + 0.5f / 128.f;
  red[i] = res; __syncthreads();
  float c = 0.f;
  for (int j = 0; j <= i; j++) c += red[j];
  cdf[i] = c; __syncthreads();

  float u = u01((unsigned)(((t * 32 + b) * 128 + i) ^ 0x5bd1e995u)) * cdf[127];
  int idx = 127;
  for (int j = 0; j < 128; j++) { if (cdf[j] >= u) { idx = j; break; } }
  int src = idx * 32 + b;
  const float4* __restrict__ s4 = (const float4*)(h1f + (size_t)src * 128);
  v4h* __restrict__ d4 = (v4h*)(h16 + (size_t)row * 128);
  for (int k4 = 0; k4 < 32; k4++) {
    float4 v = s4[k4];
    v4h hv = { (_Float16)v.x, (_Float16)v.y, (_Float16)v.z, (_Float16)v.w };
    d4[k4] = hv;
  }

  float pe = expf(p1s[idx]);
  float pn = logf(pe / (0.5f * pe + 0.5f / 128.f));
  __syncthreads();
  red[i] = pn; __syncthreads();
  for (int st = 64; st > 0; st >>= 1) {
    if (i < st) red[i] = fmaxf(red[i], red[i + st]);
    __syncthreads();
  }
  float mx2 = red[0]; __syncthreads();
  red[i] = expf(pn - mx2); __syncthreads();
  for (int st = 64; st > 0; st >>= 1) {
    if (i < st) red[i] += red[i + st];
    __syncthreads();
  }
  p[row] = pn - (mx2 + logf(red[0]));
}

// ---------- per-step output head: y[t] (weighted mean) and pf[t] ----------
__global__ void k_out(int t, const _Float16* __restrict__ h16, const float* __restrict__ p,
                      const float* __restrict__ h2l_w, const float* __restrict__ h2l_b,
                      float* __restrict__ out) {
  __shared__ float sv[128];
  int bx = blockIdx.x;
  if (bx < 32) {
    int b = bx, k = threadIdx.x;
    float acc = 0.f;
    for (int i = 0; i < 128; i++)
      acc += expf(p[i * 32 + b]) * (float)h16[(size_t)(i * 32 + b) * 128 + k];
    sv[k] = acc; __syncthreads();
    if (threadIdx.x < 3) {
      int j = threadIdx.x;
      float a = h2l_b[j];
      for (int k2 = 0; k2 < 128; k2++) a += sv[k2] * h2l_w[j * 128 + k2];
      out[(size_t)(t * 32 + b) * 3 + j] = sigm(a);
    }
  } else {
    int r = bx - 32;
    sv[threadIdx.x] = (float)h16[(size_t)r * 128 + threadIdx.x];
    __syncthreads();
    if (threadIdx.x < 3) {
      int j = threadIdx.x;
      float a = h2l_b[j];
      for (int k = 0; k < 128; k++) a += sv[k] * h2l_w[j * 128 + k];
      out[9600 + ((size_t)t * 4096 + r) * 3 + j] = sigm(a);
    }
  }
}

// ---------- host launcher ----------
extern "C" void kernel_launch(void* const* d_in, const int* in_sizes, int n_in,
                              void* d_out, int out_size, void* d_ws, size_t ws_size,
                              hipStream_t stream) {
  const float* map_in = (const float*)d_in[0];
  const float* obs_in = (const float*)d_in[1];
  const float* act_in = (const float*)d_in[2];
  const float* c1_w = (const float*)d_in[3];
  const float* c1_g = (const float*)d_in[4];
  const float* c1_b = (const float*)d_in[5];
  const float* c2_w = (const float*)d_in[6];
  const float* c2_g = (const float*)d_in[7];
  const float* c2_b = (const float*)d_in[8];
  const float* c3_w = (const float*)d_in[9];
  const float* c3_g = (const float*)d_in[10];
  const float* c3_b = (const float*)d_in[11];
  const float* map_w = (const float*)d_in[12];
  const float* map_b = (const float*)d_in[13];
  const float* m2o_w = (const float*)d_in[14];
  const float* m2o_b = (const float*)d_in[15];
  const float* m2a_w = (const float*)d_in[16];
  const float* m2a_b = (const float*)d_in[17];
  const float* obs_w = (const float*)d_in[18];
  const float* obs_b = (const float*)d_in[19];
  const float* act_w = (const float*)d_in[20];
  const float* act_b = (const float*)d_in[21];
  const float* fcz_w = (const float*)d_in[22];
  const float* fcz_b = (const float*)d_in[23];
  const float* fcr_w = (const float*)d_in[24];
  const float* fcr_b = (const float*)d_in[25];
  const float* fcn_w = (const float*)d_in[26];
  const float* fcn_b = (const float*)d_in[27];
  const float* fco_w = (const float*)d_in[28];
  const float* fco_b = (const float*)d_in[29];
  const float* h2l_w = (const float*)d_in[30];
  const float* h2l_b = (const float*)d_in[31];
  float* out = (float*)d_out;

  char* w = (char*)d_ws;
  size_t o = 0;
  auto A = [&](size_t nbytes) { size_t r = o; o += (nbytes + 255) & ~(size_t)255; return r; };
  size_t o_x1   = A(32 * 16 * 1024 * 4);
  size_t o_x2   = A(32 * 32 * 1024 * 4);
  size_t o_x3   = A(32 * 32 * 1024 * 4);
  size_t o_bnm  = A(64 * 4);
  size_t o_emap = A(32 * 128 * 4);
  size_t o_omap = A(32 * 64 * 4);
  size_t o_amap = A(32 * 64 * 4);
  size_t o_emb  = A(3200 * 128 * 2);
  size_t o_Wx   = A(128 * XS * 2);
  size_t o_xb   = A(XS * 4);
  size_t o_xp   = A((size_t)3200 * XS * 4);
  size_t o_Wzh  = A(128 * 128 * 2);
  size_t o_Wrh  = A(128 * 128 * 2);
  size_t o_Wna  = A(128 * 256 * 2);
  size_t o_Wo   = A(128 * 4);
  size_t o_h16  = A((size_t)PB * 128 * 2);
  size_t o_h1   = A((size_t)PB * 128 * 4);
  size_t o_p    = A(PB * 4);
  size_t o_lp   = A(PB * 4);

  float*     x1p   = (float*)(w + o_x1);
  float*     x2p   = (float*)(w + o_x2);
  float*     x3p   = (float*)(w + o_x3);
  float*     bnm   = (float*)(w + o_bnm);
  float*     emap  = (float*)(w + o_emap);
  float*     omap  = (float*)(w + o_omap);
  float*     amap  = (float*)(w + o_amap);
  _Float16*  embp  = (_Float16*)(w + o_emb);
  _Float16*  Wxp   = (_Float16*)(w + o_Wx);
  float*     xbp   = (float*)(w + o_xb);
  float*     xpp   = (float*)(w + o_xp);
  _Float16*  Wzhp  = (_Float16*)(w + o_Wzh);
  _Float16*  Wrhp  = (_Float16*)(w + o_Wrh);
  _Float16*  Wnap  = (_Float16*)(w + o_Wna);
  float*     Wop   = (float*)(w + o_Wo);
  _Float16*  hp    = (_Float16*)(w + o_h16);
  float*     h1p   = (float*)(w + o_h1);
  float*     pp    = (float*)(w + o_p);
  float*     lpp   = (float*)(w + o_lp);

  k_prep<<<264, 256, 0, stream>>>(fcz_w, fcz_b, fcr_w, fcr_b, fcn_w, fcn_b,
                                  fco_w, fco_b, Wzhp, Wrhp, Wnap, Wop, Wxp, xbp);

  k_conv1<<<2048, 256, 0, stream>>>(map_in, c1_w, x1p);
  k_bnstat<<<16, 256, 0, stream>>>(x1p, bnm, 16);
  k_bnapply<<<2048, 256, 0, stream>>>(x1p, bnm, c1_g, c1_b, 16);
  k_conv3x3<<<4096, 256, 0, stream>>>(x1p, c2_w, x2p, 16, 32);
  k_bnstat<<<32, 256, 0, stream>>>(x2p, bnm, 32);
  k_bnapply<<<4096, 256, 0, stream>>>(x2p, bnm, c2_g, c2_b, 32);
  k_conv3x3<<<4096, 256, 0, stream>>>(x2p, c3_w, x3p, 32, 32);
  k_bnstat<<<32, 256, 0, stream>>>(x3p, bnm, 32);
  k_bnapply<<<4096, 256, 0, stream>>>(x3p, bnm, c3_g, c3_b, 32);
  k_maplin<<<4096, 256, 0, stream>>>(x3p, map_w, map_b, emap);
  k_m2<<<32, 128, 0, stream>>>(emap, m2o_w, m2o_b, m2a_w, m2a_b, omap, amap);

  k_emb<<<1600, 256, 0, stream>>>(obs_in, act_in, obs_w, obs_b, act_w, act_b,
                                  omap, amap, embp);
  k_xpart<<<25, 256, 0, stream>>>(embp, Wxp, xbp, xpp);
  k_init<<<2048, 256, 0, stream>>>(hp, pp);

  for (int t = 0; t < S_N; t++) {
    k_stepA<<<32, 256, 0, stream>>>(t, hp, pp, xpp, Wzhp, Wrhp, Wnap, Wop, h1p, lpp);
    k_stepB<<<32, 128, 0, stream>>>(t, h1p, lpp, hp, pp);
    k_out<<<4128, 128, 0, stream>>>(t, hp, pp, h2l_w, h2l_b, out);
  }
}
